// RRG_81303730913650
// MI455X (gfx1250) — matrix operand order repaired and checked by reading, not compiled, not run on hardware
//
#include <hip/hip_runtime.h>

typedef unsigned short u16;
typedef __attribute__((ext_vector_type(16))) __bf16 v16bf;
typedef __attribute__((ext_vector_type(8)))  float  v8f;

#define NP    8192        // total points (B*N)
#define NB    4096        // points per batch
#define KNN   16
#define MOUT  256
#define KSTRIDE 40        // LDS row stride in bf16 elems (80B: 16B-aligned, de-conflicted)

__device__ __forceinline__ u16 f2bf(float f) {
    union { float f; unsigned int u; } a; a.f = f;
    unsigned int r = a.u + 0x7FFFu + ((a.u >> 16) & 1u);
    return (u16)(r >> 16);
}
__device__ __forceinline__ float bf2f(u16 h) {
    union { unsigned int u; float f; } a; a.u = ((unsigned int)h) << 16;
    return a.f;
}

union Frag {
    v16bf v;
    u16   u[16];
    uint4 q[2];
};

__device__ __forceinline__ v8f wmma_bf16(const Frag& a, const Frag& b, v8f c) {
    return __builtin_amdgcn_wmma_f32_16x16x32_bf16(
        false, a.v, false, b.v, (short)0, c, false, false);
}

__device__ __forceinline__ void load32B(Frag& f, const u16* p) {
    const uint4* q = (const uint4*)p;
    f.q[0] = q[0]; f.q[1] = q[1];
}
__device__ __forceinline__ void load2x16B(Frag& f, const u16* p0, const u16* p1) {
    f.q[0] = *(const uint4*)p0;
    f.q[1] = *(const uint4*)p1;
}

// ---------------------------------------------------------------------------
// Weight prep: W[K][M] f32 row-major -> Wt[M][K] bf16 (transposed)
// ---------------------------------------------------------------------------
__global__ void wt_kernel(const float* __restrict__ w, u16* __restrict__ wt,
                          int K, int M) {
    int i = blockIdx.x * 256 + threadIdx.x;
    int total = K * M;
    if (i >= total) return;
    int m = i / K, k = i % K;
    wt[i] = f2bf(w[(size_t)k * M + m]);
}

// Edge weight with half-padding: W[2C][M] -> Wt[M][2*Cp], each half C->Cp zero-pad
__global__ void wt_edge_kernel(const float* __restrict__ w, u16* __restrict__ wt,
                               int C, int Cp, int M) {
    int i = blockIdx.x * 256 + threadIdx.x;
    int total = M * 2 * Cp;
    if (i >= total) return;
    int m  = i / (2 * Cp);
    int kk = i % (2 * Cp);
    int half = kk >= Cp ? 1 : 0;
    int c = kk - half * Cp;
    float v = (c < C) ? w[(size_t)(half * C + c) * M + m] : 0.0f;
    wt[i] = f2bf(v);
}

// ---------------------------------------------------------------------------
// coords MLP (3->256->256) + concat [h|feat|joint] -> xcat bf16 [NP][800]
// ---------------------------------------------------------------------------
__global__ void coords_kernel(const float* __restrict__ coords,
                              const float* __restrict__ feat,
                              const float* __restrict__ joint,
                              const float* __restrict__ w1, const float* __restrict__ b1,
                              const float* __restrict__ w2, const float* __restrict__ b2,
                              u16* __restrict__ xcat) {
    __shared__ float h1[256];
    int p = blockIdx.x;
    int c = threadIdx.x;
    float c0 = coords[p * 3 + 0], c1 = coords[p * 3 + 1], c2 = coords[p * 3 + 2];
    float s = b1[c] + c0 * w1[c] + c1 * w1[256 + c] + c2 * w1[512 + c];
    h1[c] = fmaxf(s, 0.0f);
    __syncthreads();
    float s2 = b2[c];
    #pragma unroll 4
    for (int i = 0; i < 256; ++i) s2 += h1[i] * w2[i * 256 + c];
    u16* row = xcat + (size_t)p * 800;
    row[c]       = f2bf(fmaxf(s2, 0.0f));
    row[256 + c] = f2bf(feat[(size_t)p * 512 + c]);
    row[512 + c] = f2bf(feat[(size_t)p * 512 + 256 + c]);
    if (c < 16) {
        row[768 + c] = f2bf(joint[(size_t)p * 16 + c]);
        row[784 + c] = 0;   // pad 784 -> 800
    }
}

// ---------------------------------------------------------------------------
// per-row sum of squares (bf16 rows, zero-padded) -> f32
// ---------------------------------------------------------------------------
__global__ void sq_kernel(const u16* __restrict__ x, float* __restrict__ sq, int Cp) {
    int wave = threadIdx.x >> 5, lane = threadIdx.x & 31;
    int p = blockIdx.x * 8 + wave;
    const u16* row = x + (size_t)p * Cp;
    float s = 0.0f;
    for (int c = lane; c < Cp; c += 32) { float f = bf2f(row[c]); s += f * f; }
    #pragma unroll
    for (int off = 16; off > 0; off >>= 1) s += __shfl_xor(s, off);
    if (lane == 0) sq[p] = s;
}

// ---------------------------------------------------------------------------
// kNN: 16 queries per block. WMMA Gram tiles; per-wave top-16 lists in LDS;
// wave-parallel merge. idx output holds GLOBAL row indices.
// ---------------------------------------------------------------------------
__global__ void knn_kernel(const u16* __restrict__ x, const float* __restrict__ sq,
                           int* __restrict__ idx, int Cp) {
    __shared__ float qsq[16];
    __shared__ float wdist[8][16][16];
    __shared__ int   widx [8][16][16];
    __shared__ float tile [8][16][16];

    int blk = blockIdx.x;
    int qbase = blk * 16;
    int b = qbase / NB;
    int cbase = b * NB;
    int wave = threadIdx.x >> 5, lane = threadIdx.x & 31;
    int half = lane >> 4, hm = lane & 15;

    for (int t = threadIdx.x; t < 8 * 16 * 16; t += 256) {
        ((float*)wdist)[t] = 1e30f;
        ((int*)widx)[t] = cbase;
    }
    if (threadIdx.x < 16) qsq[threadIdx.x] = sq[qbase + threadIdx.x];
    __syncthreads();

    const u16* arow = x + (size_t)(qbase + hm) * Cp;
    int nsteps = Cp / 32;
    int chunks = NB / 16;                 // 256, divisible by 8 waves

    for (int ch = wave; ch < chunks; ch += 8) {
        int nbase = cbase + ch * 16;
        const u16* brow = x + (size_t)(nbase + hm) * Cp;
        v8f acc = {0.f,0.f,0.f,0.f,0.f,0.f,0.f,0.f};
        for (int s = 0; s < nsteps; ++s) {
            Frag A, B;
            const u16* ap = arow + s * 32 + half * 16;
            load32B(A, ap);
            load32B(B, brow + s * 32 + half * 16);
            __builtin_prefetch(brow + (s + 2) * 32, 0, 1);
            acc = wmma_bf16(A, B, acc);
        }
        float sc = sq[nbase + hm];
        #pragma unroll
        for (int r = 0; r < 8; ++r) {
            int m = r + 8 * half;
            tile[wave][m][hm] = qsq[m] + sc - 2.0f * acc[r];
        }
        __syncthreads();
        if (lane < 16) {
            int m = lane;
            for (int j = 0; j < 16; ++j) {
                float d = tile[wave][m][j];
                float wmax = -1e30f; int wpos = 0;
                #pragma unroll
                for (int t = 0; t < 16; ++t) {
                    float v = wdist[wave][m][t];
                    if (v > wmax) { wmax = v; wpos = t; }
                }
                if (d < wmax) {
                    wdist[wave][m][wpos] = d;
                    widx [wave][m][wpos] = nbase + j;
                }
            }
        }
        __syncthreads();
    }

    // merge 8 wave-lists (128 entries) per row -> 16 smallest; 2 rows per wave
    for (int rr = 0; rr < 2; ++rr) {
        int m = wave * 2 + rr;
        for (int t = 0; t < 16; ++t) {
            float best = 1e30f; int bw = 0, bi = 0;
            for (int f = lane; f < 128; f += 32) {
                int w = f >> 4, i = f & 15;
                float d = wdist[w][m][i];
                if (d < best) { best = d; bw = w; bi = i; }
            }
            #pragma unroll
            for (int off = 16; off > 0; off >>= 1) {
                float ob  = __shfl_xor(best, off);
                int   obw = __shfl_xor(bw, off);
                int   obi = __shfl_xor(bi, off);
                if (ob < best) { best = ob; bw = obw; bi = obi; }
            }
            if (lane == 0) {
                idx[(size_t)(qbase + m) * KNN + t] = widx[bw][m][bi];
                wdist[bw][m][bi] = 1e30f;
            }
            __threadfence_block();
        }
    }
}

// ---------------------------------------------------------------------------
// EdgeConv, 8 points per block (one per wave). Weight K-slabs (256ch x 32k)
// are staged cooperatively into LDS once per K-step and shared by all 8
// points -> 8x less L2 weight traffic. Each wave keeps 16 accumulator tiles
// (all 256 output channels) and issues 16 independent WMMAs per K-step.
// A = [x_i | x_j - x_i] (16 edges rows), relu + max over edges (M dim).
// ---------------------------------------------------------------------------
__global__ void edge_kernel(const u16* __restrict__ x, const int* __restrict__ idx,
                            const u16* __restrict__ wt, const float* __restrict__ bias,
                            u16* __restrict__ out, int Cp) {
    __shared__ u16 wslab[256 * KSTRIDE];   // 20KB: 256 channels x 32 k (padded rows)

    int wave = threadIdx.x >> 5, lane = threadIdx.x & 31;
    int half = lane >> 4, hm = lane & 15;
    int p = blockIdx.x * 8 + wave;

    int nbr = idx[(size_t)p * KNN + hm];
    const u16* xi = x + (size_t)p   * Cp;
    const u16* xj = x + (size_t)nbr * Cp;

    int h = Cp / 32;
    int nsteps2 = 2 * h;
    size_t wstride = (size_t)2 * Cp;
    const u16* wrow = wt + (size_t)threadIdx.x * wstride;  // channel this thread stages

    v8f zero = {0.f,0.f,0.f,0.f,0.f,0.f,0.f,0.f};
    v8f acc[16];
    #pragma unroll
    for (int i = 0; i < 16; ++i) acc[i] = zero;

    for (int s = 0; s < nsteps2; ++s) {
        // --- cooperative stage: thread t brings 64B of channel t's k-slab ---
        const uint4* g = (const uint4*)(wrow + s * 32);
        uint4 q0 = g[0], q1 = g[1], q2 = g[2], q3 = g[3];
        __builtin_prefetch(wrow + (s + 1) * 32, 0, 1);
        __syncthreads();                       // previous slab fully consumed
        uint4* l = (uint4*)&wslab[threadIdx.x * KSTRIDE];
        l[0] = q0; l[1] = q1; l[2] = q2; l[3] = q3;
        __syncthreads();                       // slab visible to all waves

        // --- A fragment for this wave's point ---
        Frag A;
        if (s < h) {
            const u16* ap = xi + s * 32 + half * 8;
            load2x16B(A, ap, ap + 16);
        } else {
            int kb = (s - h) * 32 + half * 8;
            #pragma unroll
            for (int g2 = 0; g2 < 2; ++g2)
                #pragma unroll
                for (int i = 0; i < 8; ++i) {
                    int c = kb + g2 * 16 + i;
                    A.u[g2 * 8 + i] = f2bf(bf2f(xj[c]) - bf2f(xi[c]));
                }
        }

        // --- 16 WMMAs over all 256 output channels from LDS slab ---
        #pragma unroll
        for (int nc = 0; nc < 8; ++nc) {
            Frag B0, B1;
            load2x16B(B0, &wslab[(nc * 32 + hm)      * KSTRIDE + half * 8], &wslab[(nc * 32 + hm)      * KSTRIDE + half * 8 + 16]);
            load2x16B(B1, &wslab[(nc * 32 + 16 + hm) * KSTRIDE + half * 8], &wslab[(nc * 32 + 16 + hm) * KSTRIDE + half * 8 + 16]);
            acc[2 * nc]     = wmma_bf16(A, B0, acc[2 * nc]);
            acc[2 * nc + 1] = wmma_bf16(A, B1, acc[2 * nc + 1]);
        }
    }

    // --- epilogue: relu + max over 16 edges (WMMA M dimension) ---
    #pragma unroll
    for (int a = 0; a < 16; ++a) {
        int n = (a >> 1) * 32 + (a & 1) * 16 + hm;
        float mx = acc[a][0];
        #pragma unroll
        for (int r = 1; r < 8; ++r) mx = fmaxf(mx, acc[a][r]);
        mx = fmaxf(mx, __shfl_xor(mx, 16));    // combine M 0-7 with 8-15
        float o = fmaxf(mx + bias[n], 0.0f);
        if (half == 0) out[(size_t)p * MOUT + n] = f2bf(o);
    }
}

// ---------------------------------------------------------------------------
// Dense layer: out = relu(X @ Wt^T + b), X bf16 [NP][Kp], Wt bf16 [256][Kp]
// ---------------------------------------------------------------------------
__global__ void dense_kernel(const u16* __restrict__ x, const u16* __restrict__ wt,
                             const float* __restrict__ bias, u16* __restrict__ out,
                             int Kp) {
    int qbase = blockIdx.x * 16;
    int wave = threadIdx.x >> 5, lane = threadIdx.x & 31;
    int half = lane >> 4, hm = lane & 15;
    int n0 = wave * 32;

    const u16* arow = x + (size_t)(qbase + hm) * Kp;
    int nsteps = Kp / 32;
    v8f acc0 = {0.f,0.f,0.f,0.f,0.f,0.f,0.f,0.f};
    v8f acc1 = {0.f,0.f,0.f,0.f,0.f,0.f,0.f,0.f};

    for (int s = 0; s < nsteps; ++s) {
        Frag A, B0, B1;
        const u16* ap = arow + s * 32 + half * 16;
        load32B(A, ap);
        const u16* wr0 = wt + (size_t)(n0 + hm) * Kp + s * 32 + half * 16;
        load32B(B0, wr0);
        load32B(B1, wr0 + (size_t)16 * Kp);
        acc0 = wmma_bf16(A, B0, acc0);
        acc1 = wmma_bf16(A, B1, acc1);
    }

    #pragma unroll
    for (int cc = 0; cc < 2; ++cc) {
        v8f a = cc ? acc1 : acc0;
        int n = n0 + cc * 16 + hm;
        float bv = bias[n];
        #pragma unroll
        for (int r = 0; r < 8; ++r) {
            int row = qbase + r + 8 * half;
            out[(size_t)row * MOUT + n] = f2bf(fmaxf(a[r] + bv, 0.0f));
        }
    }
}

// ---------------------------------------------------------------------------
// residual add (bf16)
// ---------------------------------------------------------------------------
__global__ void add_kernel(const u16* __restrict__ a, const u16* __restrict__ b,
                           u16* __restrict__ o, int total) {
    int i = blockIdx.x * 256 + threadIdx.x;
    if (i < total) o[i] = f2bf(bf2f(a[i]) + bf2f(b[i]));
}

// ---------------------------------------------------------------------------
// head: out = relu(t @ w + b), t bf16 [NP][256], w f32 [256][O], O in {3,1}
// ---------------------------------------------------------------------------
__global__ void head_kernel(const u16* __restrict__ t, const float* __restrict__ w,
                            const float* __restrict__ b, float* __restrict__ out, int O) {
    int p = blockIdx.x;
    int lane = threadIdx.x;
    const u16* row = t + (size_t)p * MOUT;
    for (int o = 0; o < O; ++o) {
        float s = 0.0f;
        for (int c = lane; c < MOUT; c += 32) s += bf2f(row[c]) * w[c * O + o];
        #pragma unroll
        for (int off = 16; off > 0; off >>= 1) s += __shfl_xor(s, off);
        if (lane == 0) out[(size_t)p * O + o] = fmaxf(s + b[o], 0.0f);
    }
}

// ---------------------------------------------------------------------------
extern "C" void kernel_launch(void* const* d_in, const int* in_sizes, int n_in,
                              void* d_out, int out_size, void* d_ws, size_t ws_size,
                              hipStream_t stream) {
    (void)in_sizes; (void)n_in; (void)out_size; (void)ws_size;

    const float* coords = (const float*)d_in[0];
    const float* feat   = (const float*)d_in[1];
    const float* joint  = (const float*)d_in[2];
    const float* w_h1 = (const float*)d_in[3];  const float* b_h1 = (const float*)d_in[4];
    const float* w_h2 = (const float*)d_in[5];  const float* b_h2 = (const float*)d_in[6];
    const float* w_e1 = (const float*)d_in[7];  const float* b_e1 = (const float*)d_in[8];
    const float* w_e2 = (const float*)d_in[9];  const float* b_e2 = (const float*)d_in[10];
    const float* w_h3 = (const float*)d_in[11]; const float* b_h3 = (const float*)d_in[12];
    const float* w_c1 = (const float*)d_in[13]; const float* b_c1 = (const float*)d_in[14];
    const float* w_c2 = (const float*)d_in[15]; const float* b_c2 = (const float*)d_in[16];
    const float* w_c3 = (const float*)d_in[17]; const float* b_c3 = (const float*)d_in[18];
    const float* w_h4 = (const float*)d_in[19]; const float* b_h4 = (const float*)d_in[20];
    const float* w_o1 = (const float*)d_in[21]; const float* b_o1 = (const float*)d_in[22];
    const float* w_h5 = (const float*)d_in[23]; const float* b_h5 = (const float*)d_in[24];
    const float* w_o2 = (const float*)d_in[25]; const float* b_o2 = (const float*)d_in[26];

    char* wp = (char*)d_ws;
    auto carve = [&](size_t n) -> char* {
        char* r = wp; wp += (n + 255) & ~(size_t)255; return r;
    };

    const int Cp0 = 800;   // 784 padded to multiple of 32
    u16*   xcat = (u16*)  carve((size_t)NP * Cp0 * 2);
    float* sqb  = (float*)carve((size_t)NP * 4);
    int*   idxb = (int*)  carve((size_t)NP * KNN * 4);
    u16* y1  = (u16*)carve((size_t)NP * MOUT * 2);
    u16* y2  = (u16*)carve((size_t)NP * MOUT * 2);
    u16* y3  = (u16*)carve((size_t)NP * MOUT * 2);
    u16* e1b = (u16*)carve((size_t)NP * MOUT * 2);
    u16* e2b = (u16*)carve((size_t)NP * MOUT * 2);
    u16* xa  = (u16*)carve((size_t)NP * MOUT * 2);
    u16* e3b = (u16*)carve((size_t)NP * MOUT * 2);
    u16* xb  = (u16*)carve((size_t)NP * MOUT * 2);
    u16* t4  = (u16*)carve((size_t)NP * MOUT * 2);
    u16* t5  = (u16*)carve((size_t)NP * MOUT * 2);
    u16* we1t = (u16*)carve((size_t)256 * 1600 * 2);
    u16* we2t = (u16*)carve((size_t)256 * 512 * 2);
    u16* wc1t = (u16*)carve((size_t)256 * 512 * 2);
    u16* wc2t = (u16*)carve((size_t)256 * 512 * 2);
    u16* wc3t = (u16*)carve((size_t)256 * 512 * 2);
    u16* wh3t = (u16*)carve((size_t)256 * 256 * 2);
    u16* wh4t = (u16*)carve((size_t)256 * 256 * 2);
    u16* wh5t = (u16*)carve((size_t)256 * 256 * 2);

    // ---- weight prep (bf16, transposed, padded) ----
    wt_edge_kernel<<<(256 * 1600 + 255) / 256, 256, 0, stream>>>(w_e1, we1t, 784, 800, 256);
    wt_kernel<<<(512 * 256 + 255) / 256, 256, 0, stream>>>(w_e2, we2t, 512, 256);
    wt_kernel<<<(512 * 256 + 255) / 256, 256, 0, stream>>>(w_c1, wc1t, 512, 256);
    wt_kernel<<<(512 * 256 + 255) / 256, 256, 0, stream>>>(w_c2, wc2t, 512, 256);
    wt_kernel<<<(512 * 256 + 255) / 256, 256, 0, stream>>>(w_c3, wc3t, 512, 256);
    wt_kernel<<<(256 * 256 + 255) / 256, 256, 0, stream>>>(w_h3, wh3t, 256, 256);
    wt_kernel<<<(256 * 256 + 255) / 256, 256, 0, stream>>>(w_h4, wh4t, 256, 256);
    wt_kernel<<<(256 * 256 + 255) / 256, 256, 0, stream>>>(w_h5, wh5t, 256, 256);

    // ---- stem: coords MLP + concat ----
    coords_kernel<<<NP, 256, 0, stream>>>(coords, feat, joint, w_h1, b_h1, w_h2, b_h2, xcat);

    // ---- EdgeConvE 1 (C=784/800) ----
    sq_kernel<<<NP / 8, 256, 0, stream>>>(xcat, sqb, Cp0);
    knn_kernel<<<NP / 16, 256, 0, stream>>>(xcat, sqb, idxb, Cp0);
    edge_kernel<<<NP / 8, 256, 0, stream>>>(xcat, idxb, we1t, b_e1, y1, Cp0);

    // ---- EdgeConvE 2 (C=256) ----
    sq_kernel<<<NP / 8, 256, 0, stream>>>(y1, sqb, MOUT);
    knn_kernel<<<NP / 16, 256, 0, stream>>>(y1, sqb, idxb, MOUT);
    edge_kernel<<<NP / 8, 256, 0, stream>>>(y1, idxb, we2t, b_e2, y2, MOUT);

    // ---- h3 dense ----
    dense_kernel<<<NP / 16, 256, 0, stream>>>(y2, wh3t, b_h3, y3, MOUT);

    // ---- EdgeConv c1 ----
    sq_kernel<<<NP / 8, 256, 0, stream>>>(y3, sqb, MOUT);
    knn_kernel<<<NP / 16, 256, 0, stream>>>(y3, sqb, idxb, MOUT);
    edge_kernel<<<NP / 8, 256, 0, stream>>>(y3, idxb, wc1t, b_c1, e1b, MOUT);

    // ---- EdgeConv c2 + residual ----
    sq_kernel<<<NP / 8, 256, 0, stream>>>(e1b, sqb, MOUT);
    knn_kernel<<<NP / 16, 256, 0, stream>>>(e1b, sqb, idxb, MOUT);
    edge_kernel<<<NP / 8, 256, 0, stream>>>(e1b, idxb, wc2t, b_c2, e2b, MOUT);
    add_kernel<<<NP, 256, 0, stream>>>(e2b, e1b, xa, NP * MOUT);

    // ---- EdgeConv c3 + residual ----
    sq_kernel<<<NP / 8, 256, 0, stream>>>(xa, sqb, MOUT);
    knn_kernel<<<NP / 16, 256, 0, stream>>>(xa, sqb, idxb, MOUT);
    edge_kernel<<<NP / 8, 256, 0, stream>>>(xa, idxb, wc3t, b_c3, e3b, MOUT);
    add_kernel<<<NP, 256, 0, stream>>>(e3b, e2b, xb, NP * MOUT);

    // ---- heads ----
    float* out1 = (float*)d_out;
    float* out2 = out1 + (size_t)NP * 3;
    dense_kernel<<<NP / 16, 256, 0, stream>>>(xb, wh4t, b_h4, t4, MOUT);
    head_kernel<<<NP, 32, 0, stream>>>(t4, w_o1, b_o1, out1, 3);
    dense_kernel<<<NP / 16, 256, 0, stream>>>(xb, wh5t, b_h5, t5, MOUT);
    head_kernel<<<NP, 32, 0, stream>>>(t5, w_o2, b_o2, out2, 1);
}
